// MyModel_43971875176991
// MI455X (gfx1250) — matrix operand order repaired and checked by reading, not compiled, not run on hardware
//
#include <hip/hip_runtime.h>
#include <hip/hip_bf16.h>

typedef __attribute__((ext_vector_type(16))) _Float16 v16h;
typedef __attribute__((ext_vector_type(8)))  float    v8f;

#define WMMA_F32_F16(a, b, c) \
  __builtin_amdgcn_wmma_f32_16x16x32_f16(false, (a), false, (b), (short)0, (c), false, false)

constexpr int cB = 32, cS = 512, cE = 256, cH = 8, cD = 32, cV = 30522, cHID = 256, cL = 4;

// A-fragment (16x32 f16) element -> K index: contiguous order k = 16*half + e,
// matching the B-fragment fills; LDS fragment reads merge into 2x ds_load_b128.
__device__ __forceinline__ int afrag_k(int e, int half) {
  return half * 16 + e;
}

// ---------------------------------------------------------------------------
// WMMA GEMM: C[M,N] = A[M,K] @ B[K,N] (+bias, +gelu), f32 in/out, f16 compute.
// LDA/NN/KK are compile-time so all global loads are base+immediate-offset.
// Block = 256 threads (8 waves). Block tile = 16(M) x 256(N); each wave owns
// two 16x16 WMMA tiles (cols wv*16 and wv*16+128). Double-buffered LDS with
// register staging; K-loop fully unrolled. flags: bit0 = +bias, bit1 = GELU.
// ---------------------------------------------------------------------------
template<int LDA, int NN, int KK>
__global__ __launch_bounds__(256) void gemm_wmma(
    const float* __restrict__ A,
    const float* __restrict__ Bw,
    const float* __restrict__ bias,
    float* __restrict__ C,
    int flags)
{
  constexpr int  NK     = KK / 32;
  constexpr bool AL16   = (NN % 4 == 0);     // rows of B 16B-aligned -> float4
  constexpr bool CANRAG = (NN % 256 != 0);   // only the logits GEMM is ragged

  __shared__ _Float16 Alds[2][16][40];       // row stride 80B -> b128-aligned
  __shared__ _Float16 Blds[2][256][40];      // [n][k], row stride 80B
  const int tid  = threadIdx.x;
  const int wv   = tid >> 5, lane = tid & 31;
  const int half = lane >> 4, ln16 = lane & 15;
  const int row0 = blockIdx.y * 16;
  const int col0 = blockIdx.x * 256;
  const bool ragged = CANRAG && (col0 + 256 > NN);   // block-uniform

  // A staging: thread -> (row ar, cols ak..ak+1), one float2
  const int ar = tid >> 4, ak = (tid & 15) * 2;
  const float* aptr = A + (row0 + ar) * LDA + ak;
  // B staging map (AL16: float4 x8 rows; else float2 x16 rows)
  const int bn = AL16 ? (tid & 63) * 4 : (tid & 127) * 2;
  const int bk = AL16 ? (tid >> 6) * 8 : (tid >> 7) * 16;
  const float* bptr = Bw + bk * NN + col0 + bn;

  float2 areg;
  float4 breg[8];
  _Float16 bh0[16], bh1[16];

  auto loadTiles = [&](int kb) {
    areg = *(const float2*)(aptr + kb);
    if constexpr (AL16) {
      const float* bp = bptr + kb * NN;
      #pragma unroll
      for (int j = 0; j < 8; ++j) breg[j] = *(const float4*)(bp + j * NN);
    } else {
      if (!ragged) {
        const float* bp = bptr + kb * NN;
        #pragma unroll
        for (int j = 0; j < 16; ++j) {
          float2 v = *(const float2*)(bp + j * NN);
          bh0[j] = (_Float16)v.x;
          bh1[j] = (_Float16)v.y;
        }
      } else {
        int c0 = col0 + bn;
        #pragma unroll
        for (int j = 0; j < 16; ++j) {
          int r = (kb + bk + j) * NN;
          bh0[j] = (_Float16)((c0     < NN) ? Bw[r + c0]     : 0.f);
          bh1[j] = (_Float16)((c0 + 1 < NN) ? Bw[r + c0 + 1] : 0.f);
        }
      }
    }
  };
  auto storeTiles = [&](int buf) {
    Alds[buf][ar][ak]     = (_Float16)areg.x;
    Alds[buf][ar][ak + 1] = (_Float16)areg.y;
    if constexpr (AL16) {
      #pragma unroll
      for (int j = 0; j < 8; ++j) {          // per-column j-runs -> b128 stores
        Blds[buf][bn + 0][bk + j] = (_Float16)breg[j].x;
        Blds[buf][bn + 1][bk + j] = (_Float16)breg[j].y;
        Blds[buf][bn + 2][bk + j] = (_Float16)breg[j].z;
        Blds[buf][bn + 3][bk + j] = (_Float16)breg[j].w;
      }
    } else {
      #pragma unroll
      for (int j = 0; j < 16; ++j) {
        Blds[buf][bn][bk + j]     = bh0[j];
        Blds[buf][bn + 1][bk + j] = bh1[j];
      }
    }
  };

  v8f acc0 = {}, acc1 = {};
  loadTiles(0);
  #pragma unroll
  for (int kt = 0; kt < NK; ++kt) {
    const int buf = kt & 1;                  // constant after unroll
    storeTiles(buf);
    __syncthreads();
    if (kt + 1 < NK) loadTiles((kt + 1) * 32);   // overlaps WMMA below
    v16h af, bf0, bf1;
    #pragma unroll
    for (int e = 0; e < 16; ++e) af[e] = Alds[buf][ln16][afrag_k(e, half)];
    #pragma unroll
    for (int e = 0; e < 16; ++e) bf0[e] = Blds[buf][wv * 16 + ln16][half * 16 + e];
    #pragma unroll
    for (int e = 0; e < 16; ++e) bf1[e] = Blds[buf][128 + wv * 16 + ln16][half * 16 + e];
    acc0 = WMMA_F32_F16(af, bf0, acc0);
    acc1 = WMMA_F32_F16(af, bf1, acc1);
  }

  // Epilogue (C layout: vgpr r -> row r + 8*half, col = lane%16)
  #pragma unroll
  for (int r = 0; r < 8; ++r) {
    int row = row0 + r + 8 * half;
    #pragma unroll
    for (int t = 0; t < 2; ++t) {
      int col = col0 + t * 128 + wv * 16 + ln16;
      if (!CANRAG || col < NN) {
        float v = (t == 0) ? acc0[r] : acc1[r];
        if (flags & 1) v += bias[col];
        if (flags & 2) v = 0.5f * v * (1.f + erff(v * 0.70710678118f));
        C[row * NN + col] = v;
      }
    }
  }
}

// ---------------------------------------------------------------------------
// Fused flash-style attention (no 1/sqrt(d) scale, per reference). Block = 8
// waves, all on the same (b,h); wave w owns q-rows [qbase, qbase+16). Per
// 32-key chunk: K and V staged cooperatively in double-buffered LDS
// (coalesced float4 loads, V transposed so P@V B-frags are contiguous b128
// LDS reads), 2 WMMAs for Q@K^T, p = exp(s) directly (scores are O(1): LN'd
// activations through 0.02-scale weights -> no overflow risk in f32 exp or
// f16 p), 2 WMMAs into the O accumulators. Row sums are accumulated as
// lane-local column partials; ONE 4-step 16-lane reduction at the end
// (gfx1250 shfl == ds_bpermute, so per-chunk butterflies would swamp the DS
// pipe). Denominator sums the f16-rounded p to match the WMMA numerator.
// ---------------------------------------------------------------------------
__global__ __launch_bounds__(256) void attn_wmma(
    const float* __restrict__ Q, const float* __restrict__ Kt,
    const float* __restrict__ V, float* __restrict__ O)
{
  __shared__ _Float16 Klds[2][32][40];     // [key][dim]
  __shared__ _Float16 Vlds[2][32][40];     // [dim][key] (transposed)
  __shared__ _Float16 Plds[8][16][40];     // per-wave P tile
  const int tid  = threadIdx.x;
  const int wv   = tid >> 5, lane = tid & 31;
  const int half = lane >> 4, ln16 = lane & 15;
  const int b = blockIdx.z, h = blockIdx.y;
  const int qbase = (blockIdx.x * 8 + wv) * 16;
  const int base = (b * cS) * cE + h * cD;     // + row*cE + d  (fits int32)

  // Q fragment (A layout): row = qbase + lane%16, k = head dim
  v16h aq;
  {
    const float* qp = Q + base + (qbase + ln16) * cE;
    #pragma unroll
    for (int e = 0; e < 16; ++e) aq[e] = (_Float16)qp[afrag_k(e, half)];
  }

  // Cooperative K/V staging map: thread -> (key skey, dims sd0..sd0+3)
  const int skey = tid >> 3, sd0 = (tid & 7) * 4;
  const float* kstage = Kt + base + skey * cE + sd0;
  const float* vstage = V  + base + skey * cE + sd0;
  float4 kreg, vreg;
  auto loadKV = [&](int kc) {
    kreg = *(const float4*)(kstage + kc * cE);
    vreg = *(const float4*)(vstage + kc * cE);
  };
  auto storeKV = [&](int buf) {
    Klds[buf][skey][sd0]     = (_Float16)kreg.x;   // contiguous -> ds_store_b64
    Klds[buf][skey][sd0 + 1] = (_Float16)kreg.y;
    Klds[buf][skey][sd0 + 2] = (_Float16)kreg.z;
    Klds[buf][skey][sd0 + 3] = (_Float16)kreg.w;
    Vlds[buf][sd0][skey]     = (_Float16)vreg.x;   // transpose scatter
    Vlds[buf][sd0 + 1][skey] = (_Float16)vreg.y;
    Vlds[buf][sd0 + 2][skey] = (_Float16)vreg.z;
    Vlds[buf][sd0 + 3][skey] = (_Float16)vreg.w;
  };

  v8f acc0 = {}, acc1 = {};
  float lsum[8];
  #pragma unroll
  for (int r = 0; r < 8; ++r) lsum[r] = 0.f;

  loadKV(0);
  #pragma unroll 2                              // folds buf to constants
  for (int kc = 0; kc < cS; kc += 32) {
    const int buf = (kc >> 5) & 1;
    storeKV(buf);
    __syncthreads();
    if (kc + 32 < cS) loadKV(kc + 32);          // overlaps compute below

    // K^T fragments: column = key, k-dim = head dim (b128 LDS reads)
    v16h bk0, bk1;
    #pragma unroll
    for (int e = 0; e < 16; ++e) {
      bk0[e] = Klds[buf][ln16][half * 16 + e];
      bk1[e] = Klds[buf][16 + ln16][half * 16 + e];
    }
    v8f z = {};
    v8f s0 = WMMA_F32_F16(aq, bk0, z);
    v8f s1 = WMMA_F32_F16(aq, bk1, z);

    // p = exp(s); lane-local partial row sums (this lane's 2 columns/chunk)
    #pragma unroll
    for (int r = 0; r < 8; ++r) {
      _Float16 p0 = (_Float16)__expf(s0[r]);
      _Float16 p1 = (_Float16)__expf(s1[r]);
      lsum[r] += (float)p0 + (float)p1;
      Plds[wv][r + 8 * half][ln16]      = p0;   // per-wave staging
      Plds[wv][r + 8 * half][ln16 + 16] = p1;
    }
    asm volatile("s_wait_dscnt 0" ::: "memory");  // DS pipe is in-order per wave

    v16h ap;
    #pragma unroll
    for (int e = 0; e < 16; ++e) ap[e] = Plds[wv][ln16][afrag_k(e, half)];

    // V fragments: column = head dim, k-dim = key (b128 LDS reads)
    v16h bv0, bv1;
    #pragma unroll
    for (int e = 0; e < 16; ++e) {
      bv0[e] = Vlds[buf][ln16][half * 16 + e];
      bv1[e] = Vlds[buf][16 + ln16][half * 16 + e];
    }
    acc0 = WMMA_F32_F16(ap, bv0, acc0);
    acc1 = WMMA_F32_F16(ap, bv1, acc1);
  }

  // One cross-lane reduction per row at the end (16-lane groups)
  #pragma unroll
  for (int r = 0; r < 8; ++r) {
    #pragma unroll
    for (int off = 1; off < 16; off <<= 1)
      lsum[r] += __shfl_xor(lsum[r], off, 32);
  }

  #pragma unroll
  for (int r = 0; r < 8; ++r) {
    float inv = 1.f / lsum[r];
    int orow = base + (qbase + r + 8 * half) * cE;
    O[orow + ln16]      = acc0[r] * inv;
    O[orow + 16 + ln16] = acc1[r] * inv;
  }
}

// ---------------------------------------------------------------------------
// out = LayerNorm(a + b) * g + be  (E=256, one wave32 per row, 8 elems/lane)
// ---------------------------------------------------------------------------
__global__ __launch_bounds__(256) void add_layernorm(
    const float* __restrict__ a, const float* __restrict__ b,
    const float* __restrict__ g, const float* __restrict__ be,
    float* __restrict__ out, int rows)
{
  int wv = threadIdx.x >> 5, lane = threadIdx.x & 31;
  int row = blockIdx.x * 8 + wv;
  if (row >= rows) return;
  const float* ap = a + row * cE;
  const float* bp = b + row * cE;
  float vals[8], s = 0.f, s2 = 0.f;
  #pragma unroll
  for (int j = 0; j < 8; ++j) {
    int e = j * 32 + lane;
    float v = ap[e] + bp[e];
    vals[j] = v; s += v; s2 += v * v;
  }
  #pragma unroll
  for (int off = 1; off < 32; off <<= 1) {
    s  += __shfl_xor(s,  off, 32);
    s2 += __shfl_xor(s2, off, 32);
  }
  float mean = s * (1.f / cE);
  float var  = s2 * (1.f / cE) - mean * mean;
  float rstd = rsqrtf(var + 1e-5f);
  #pragma unroll
  for (int j = 0; j < 8; ++j) {
    int e = j * 32 + lane;
    out[row * cE + e] = (vals[j] - mean) * rstd * g[e] + be[e];
  }
}

// ---------------------------------------------------------------------------
// x[b,s,:] = emb[tokens[b,s],:] + sinusoidal PE (angle pos/10000^(i/E);
// even i -> sin, odd i -> cos, matching the reference exactly)
// ---------------------------------------------------------------------------
__global__ __launch_bounds__(256) void embed_pe(
    const int* __restrict__ tokens, const float* __restrict__ emb,
    float* __restrict__ x)
{
  int bs = blockIdx.x;
  int i = threadIdx.x;
  int s = bs % cS;
  int tok = tokens[bs];
  float ang = (float)s * __expf(-((float)i / (float)cE) * 9.210340371976184f); // ln(1e4)
  float pe = (i & 1) ? __cosf(ang) : __sinf(ang);
  x[bs * cE + i] = emb[tok * cE + i] + pe;
}

// ---------------------------------------------------------------------------
extern "C" void kernel_launch(void* const* d_in, const int* in_sizes, int n_in,
                              void* d_out, int out_size, void* d_ws, size_t ws_size,
                              hipStream_t stream)
{
  const int*   tokens = (const int*)  d_in[0];
  const float* emb    = (const float*)d_in[1];
  const float* qw     = (const float*)d_in[2];
  const float* kw     = (const float*)d_in[3];
  const float* vw     = (const float*)d_in[4];
  const float* wo     = (const float*)d_in[5];
  const float* bo     = (const float*)d_in[6];
  const float* g1     = (const float*)d_in[7];
  const float* be1    = (const float*)d_in[8];
  const float* w1     = (const float*)d_in[9];
  const float* c1     = (const float*)d_in[10];
  const float* w2     = (const float*)d_in[11];
  const float* c2     = (const float*)d_in[12];
  const float* g2     = (const float*)d_in[13];
  const float* be2    = (const float*)d_in[14];
  const float* wout   = (const float*)d_in[15];
  const float* bout   = (const float*)d_in[16];
  float* out = (float*)d_out;

  const int nTok = cB * cS;            // 16384 token rows
  const long nAct = (long)nTok * cE;   // floats per activation buffer (16 MB)
  float* x    = (float*)d_ws;
  float* q    = x + nAct;
  float* k    = q + nAct;
  float* v    = k + nAct;
  float* attn = v + nAct;              // 5 x 16 MB = 80 MB workspace

  const int M = nTok;
  dim3 blk(256);
  dim3 ggrid(1, M / 16);               // N=256 per block -> A read exactly once
  dim3 agrid(cS / (16 * 8), cH, cB);   // (4, 8, 32): one wave per 16-row q tile
  dim3 lngrid(M / 8);

  embed_pe<<<dim3((unsigned)nTok), blk, 0, stream>>>(tokens, emb, x);

  for (int l = 0; l < cL; ++l) {
    const float* qwl = qw + (long)l * cE * cE;
    const float* kwl = kw + (long)l * cE * cE;
    const float* vwl = vw + (long)l * cE * cE;
    const float* wol = wo + (long)l * cE * cE;
    const float* bol = bo + (long)l * cE;
    const float* g1l = g1 + (long)l * cE;
    const float* b1l = be1 + (long)l * cE;
    const float* w1l = w1 + (long)l * cE * cHID;
    const float* c1l = c1 + (long)l * cHID;
    const float* w2l = w2 + (long)l * cHID * cE;
    const float* c2l = c2 + (long)l * cE;
    const float* g2l = g2 + (long)l * cE;
    const float* b2l = be2 + (long)l * cE;

    gemm_wmma<cE, cE, cE><<<ggrid, blk, 0, stream>>>(x, qwl, nullptr, q, 0);
    gemm_wmma<cE, cE, cE><<<ggrid, blk, 0, stream>>>(x, kwl, nullptr, k, 0);
    gemm_wmma<cE, cE, cE><<<ggrid, blk, 0, stream>>>(x, vwl, nullptr, v, 0);
    attn_wmma<<<agrid, blk, 0, stream>>>(q, k, v, attn);
    // attn @ wo + bo -> q ; LN(q + x) -> k   (post-attn residual)
    gemm_wmma<cE, cE, cE><<<ggrid, blk, 0, stream>>>(attn, wol, bol, q, 1);
    add_layernorm<<<lngrid, blk, 0, stream>>>(q, x, g1l, b1l, k, M);
    // gelu(k @ w1 + c1) -> v ; v @ w2 + c2 -> q ; LN(q + x) -> x (block-input residual!)
    gemm_wmma<cE, cE, cE><<<ggrid, blk, 0, stream>>>(k, w1l, c1l, v, 3);
    gemm_wmma<cE, cE, cE><<<ggrid, blk, 0, stream>>>(v, w2l, c2l, q, 1);
    add_layernorm<<<lngrid, blk, 0, stream>>>(q, x, g2l, b2l, x, M);
  }

  // Logits for last token only: A = x[:, S-1, :] as strided rows [32 x 256]
  dim3 lgrid((cV + 255) / 256, cB / 16);   // (120, 2)
  gemm_wmma<cS * cE, cV, cE><<<lgrid, blk, 0, stream>>>(
      x + (cS - 1) * cE, wout, bout, out, 1);
}
